// GraphChebNet_40785009443419
// MI455X (gfx1250) — compile-verified, matrix operand order checked by reading, not run on hardware
//
#include <hip/hip_runtime.h>

// ---------------------------------------------------------------------------
// ChebNet (K=2, 3 layers) for MI455X / gfx1250.
//   agg:   tx[dst] += w[e] * h[src]      (L2-resident scatter, f32 atomics)
//   gemm:  out = h @ W0 + tx @ W1 + b    (v_wmma_f32_16x16x32_bf16, f32 accum)
// Weights are pre-swizzled into the exact WMMA B-fragment per-lane layout so
// the GEMM inner loop is: 1 A-frag build (f32->bf16) + 8 B-frag 32B loads +
// 8 WMMA per K-step per wave.  No LDS, no barriers, uniform control flow
// (EXEC all-ones as WMMA requires).
// ---------------------------------------------------------------------------

typedef __attribute__((ext_vector_type(16))) __bf16 v16bf;
typedef __attribute__((ext_vector_type(8)))  float  v8f;

#define FDIM_OUT 256

// ---------------- graph norm kernels ----------------

__global__ void zero_f32(float* __restrict__ p, int n) {
    int i = blockIdx.x * blockDim.x + threadIdx.x;
    if (i < n) p[i] = 0.0f;
}

__global__ void deg_kernel(const int* __restrict__ src, const int* __restrict__ dst,
                           float* __restrict__ deg, int E) {
    int e = blockIdx.x * blockDim.x + threadIdx.x;
    if (e >= E) return;
    int s = src[e], d = dst[e];
    if (s != d) atomicAdd(&deg[s], 1.0f);
}

__global__ void dinv_kernel(float* __restrict__ deg, int N) {
    int n = blockIdx.x * blockDim.x + threadIdx.x;
    if (n >= N) return;
    float dg = deg[n];
    deg[n] = (dg > 0.0f) ? rsqrtf(dg) : 0.0f;   // deg buffer becomes dinv
}

__global__ void edgew_kernel(const int* __restrict__ src, const int* __restrict__ dst,
                             const float* __restrict__ dinv, float* __restrict__ w, int E) {
    int e = blockIdx.x * blockDim.x + threadIdx.x;
    if (e >= E) return;
    int s = src[e], d = dst[e];
    w[e] = (s != d) ? (-dinv[s] * dinv[d]) : 0.0f;
}

// ---------------- weight pre-swizzle ----------------
// Packed layout: [K/32 kk][16 nt][32 lane] x v16bf.
// Lane l holds column n = nt*16 + (l&15); element j holds
// k = kk*32 + (l>>4)*8 + (j<8 ? j : j+8)   (ISA 7.12.2, 16-bit 32x16 B tile).
__global__ void pack_weight(const float* __restrict__ W, __bf16* __restrict__ P, int K) {
    int t = blockIdx.x * blockDim.x + threadIdx.x;
    int total = (K / 32) * 16 * 32;
    if (t >= total) return;
    int lane = t & 31;
    int nt   = (t >> 5) & 15;
    int kk   = t >> 9;
    int n    = nt * 16 + (lane & 15);
    int kb   = kk * 32 + (lane >> 4) * 8;
    v16bf v;
#pragma unroll
    for (int j = 0; j < 16; ++j) {
        int k = kb + (j < 8 ? j : j + 8);
        v[j] = (__bf16)W[(size_t)k * FDIM_OUT + n];
    }
    *((v16bf*)(P + (size_t)t * 16)) = v;
}

// ---------------- sparse aggregation ----------------
// One wave per edge; lanes stripe the feature dimension (float4 gathers,
// per-float global_atomic_add_f32 scatters into L2).
__global__ void spmm_kernel(const int* __restrict__ src, const int* __restrict__ dst,
                            const float* __restrict__ w, const float* __restrict__ X,
                            float* __restrict__ TX, int E, int F) {
    int wave = threadIdx.x >> 5;
    int lane = threadIdx.x & 31;
    int e = blockIdx.x * (blockDim.x >> 5) + wave;
    if (e >= E) return;
    float wt = w[e];
    if (wt == 0.0f) return;
    int s = src[e], d = dst[e];
    const float4* xr = (const float4*)(X + (size_t)s * F);
    float* tr = TX + (size_t)d * F;
    int nq = F >> 2;
    for (int c = lane; c < nq; c += 32) {
        float4 v = xr[c];
        atomicAdd(tr + c * 4 + 0, wt * v.x);
        atomicAdd(tr + c * 4 + 1, wt * v.y);
        atomicAdd(tr + c * 4 + 2, wt * v.z);
        atomicAdd(tr + c * 4 + 3, wt * v.w);
    }
}

// ---------------- WMMA GEMM ----------------
// out[N,256] = relu?( X[N,K] @ W0 + TX[N,K] @ W1 + b )
// Block = 256 threads = 8 waves = 4 row-tiles x 2 col-halves.
// Wave tile: 16 rows x 128 cols = 8 accumulators (v8f each).
__global__ void __launch_bounds__(256)
gemm_cheb(const float* __restrict__ X, const float* __restrict__ TX,
          const v16bf* __restrict__ Wp0, const v16bf* __restrict__ Wp1,
          const float* __restrict__ bias, float* __restrict__ out,
          int K, int rowTiles, int relu) {
    const int lane  = threadIdx.x & 31;
    const int wave  = threadIdx.x >> 5;
    const int m     = lane & 15;        // output column within 16-tile / A row
    const int khalf = lane >> 4;        // lane-half selects K sub-block
    int rowTile = blockIdx.x * 4 + (wave >> 1);
    if (rowTile >= rowTiles) rowTile = rowTiles - 1;  // tail: duplicate tile (identical data)
    const int colHalf = wave & 1;
    const int row0 = rowTile * 16;
    const int colBase = colHalf * 128;

    // init accumulators with bias (C layout: lane -> N=m, vgpr r -> M=r+8*khalf)
    v8f acc[8];
#pragma unroll
    for (int nt = 0; nt < 8; ++nt) {
        float bv = bias[colBase + nt * 16 + m];
        v8f c;
#pragma unroll
        for (int r = 0; r < 8; ++r) c[r] = bv;
        acc[nt] = c;
    }

    const int steps = K >> 5;  // K / 32
#pragma unroll
    for (int part = 0; part < 2; ++part) {
        const float* A   = part ? TX : X;
        const v16bf* Wp  = part ? Wp1 : Wp0;
        const float* arow = A + (size_t)(row0 + m) * K;
        for (int kk = 0; kk < steps; ++kk) {
            const int kb = kk * 32 + khalf * 8;
            // A fragment: row m, K = kb + {0..7, 16..23}, f32 -> bf16
            const float4* a0 = (const float4*)(arow + kb);
            const float4* a1 = (const float4*)(arow + kb + 16);
            float4 f0 = a0[0], f1 = a0[1], f2 = a1[0], f3 = a1[1];
            v16bf af;
            af[0]  = (__bf16)f0.x; af[1]  = (__bf16)f0.y; af[2]  = (__bf16)f0.z; af[3]  = (__bf16)f0.w;
            af[4]  = (__bf16)f1.x; af[5]  = (__bf16)f1.y; af[6]  = (__bf16)f1.z; af[7]  = (__bf16)f1.w;
            af[8]  = (__bf16)f2.x; af[9]  = (__bf16)f2.y; af[10] = (__bf16)f2.z; af[11] = (__bf16)f2.w;
            af[12] = (__bf16)f3.x; af[13] = (__bf16)f3.y; af[14] = (__bf16)f3.z; af[15] = (__bf16)f3.w;

            const v16bf* wstep = Wp + (size_t)kk * 16 * 32;
#pragma unroll
            for (int nt = 0; nt < 8; ++nt) {
                v16bf bf = wstep[(size_t)(colHalf * 8 + nt) * 32 + lane];
                acc[nt] = __builtin_amdgcn_wmma_f32_16x16x32_bf16(
                    /*neg_a=*/false, af, /*neg_b=*/false, bf,
                    /*c_mod=*/(short)0, acc[nt],
                    /*reuse_a=*/false, /*reuse_b=*/false);
            }
        }
    }

    // store (optionally fused ReLU); D layout: lane -> N=m, vgpr r -> M=r+8*khalf
    const int rbase = row0 + khalf * 8;
#pragma unroll
    for (int nt = 0; nt < 8; ++nt) {
        const int col = colBase + nt * 16 + m;
#pragma unroll
        for (int r = 0; r < 8; ++r) {
            float v = acc[nt][r];
            if (relu) v = fmaxf(v, 0.0f);
            out[(size_t)(rbase + r) * FDIM_OUT + col] = v;
        }
    }
}

// ---------------- host launcher ----------------

extern "C" void kernel_launch(void* const* d_in, const int* in_sizes, int n_in,
                              void* d_out, int out_size, void* d_ws, size_t ws_size,
                              hipStream_t stream) {
    const float* x    = (const float*)d_in[0];
    const int*   eidx = (const int*)d_in[1];          // [2,E] int32
    const float* W1_0 = (const float*)d_in[2];
    const float* W1_1 = (const float*)d_in[3];
    const float* b1   = (const float*)d_in[4];
    const float* W2_0 = (const float*)d_in[5];
    const float* W2_1 = (const float*)d_in[6];
    const float* b2   = (const float*)d_in[7];
    const float* W3_0 = (const float*)d_in[8];
    const float* W3_1 = (const float*)d_in[9];
    const float* b3   = (const float*)d_in[10];
    float* out = (float*)d_out;

    const int E = in_sizes[1] / 2;
    const int K1 = 128;
    const int N = in_sizes[0] / K1;                   // 50000
    const int* src = eidx;
    const int* dst = eidx + E;

    // workspace carve-out (256B aligned)
    char* ws = (char*)d_ws;
    size_t off = 0;
    auto alloc = [&](size_t bytes) -> void* {
        void* p = ws + off;
        off = (off + bytes + 255) & ~(size_t)255;
        return p;
    };
    float*  deg  = (float*)alloc((size_t)N * 4);                 // deg -> dinv
    float*  ew   = (float*)alloc((size_t)E * 4);                 // edge weights
    float*  bufT = (float*)alloc((size_t)N * FDIM_OUT * 4);      // tx (zeroed per layer)
    float*  bufA = (float*)alloc((size_t)N * FDIM_OUT * 4);      // h1
    float*  bufB = (float*)alloc((size_t)N * FDIM_OUT * 4);      // h2
    __bf16* P1_0 = (__bf16*)alloc((size_t)K1 * 256 * 2);
    __bf16* P1_1 = (__bf16*)alloc((size_t)K1 * 256 * 2);
    __bf16* P2_0 = (__bf16*)alloc((size_t)256 * 256 * 2);
    __bf16* P2_1 = (__bf16*)alloc((size_t)256 * 256 * 2);
    __bf16* P3_0 = (__bf16*)alloc((size_t)256 * 256 * 2);
    __bf16* P3_1 = (__bf16*)alloc((size_t)256 * 256 * 2);
    (void)ws_size; (void)n_in; (void)out_size;

    const int T = 256;
    auto cdiv = [](int a, int b) { return (a + b - 1) / b; };

    // ---- graph normalization ----
    zero_f32<<<cdiv(N, T), T, 0, stream>>>(deg, N);
    deg_kernel<<<cdiv(E, T), T, 0, stream>>>(src, dst, deg, E);
    dinv_kernel<<<cdiv(N, T), T, 0, stream>>>(deg, N);
    edgew_kernel<<<cdiv(E, T), T, 0, stream>>>(src, dst, deg, ew, E);

    // ---- pre-swizzle weights into WMMA fragment layout ----
    pack_weight<<<cdiv((K1 / 32) * 512, T), T, 0, stream>>>(W1_0, P1_0, K1);
    pack_weight<<<cdiv((K1 / 32) * 512, T), T, 0, stream>>>(W1_1, P1_1, K1);
    pack_weight<<<cdiv((256 / 32) * 512, T), T, 0, stream>>>(W2_0, P2_0, 256);
    pack_weight<<<cdiv((256 / 32) * 512, T), T, 0, stream>>>(W2_1, P2_1, 256);
    pack_weight<<<cdiv((256 / 32) * 512, T), T, 0, stream>>>(W3_0, P3_0, 256);
    pack_weight<<<cdiv((256 / 32) * 512, T), T, 0, stream>>>(W3_1, P3_1, 256);

    const int rowTiles  = N / 16;                 // 3125 (N % 16 == 0)
    const int gemmGrid  = cdiv(rowTiles, 4);      // 4 row-tiles per block
    const int spmmGrid  = cdiv(E, T / 32);        // one wave per edge

    // ---- layer 1: 128 -> 256, ReLU ----
    zero_f32<<<cdiv(N * K1, T), T, 0, stream>>>(bufT, N * K1);
    spmm_kernel<<<spmmGrid, T, 0, stream>>>(src, dst, ew, x, bufT, E, K1);
    gemm_cheb<<<gemmGrid, T, 0, stream>>>(x, bufT, (const v16bf*)P1_0, (const v16bf*)P1_1,
                                          b1, bufA, K1, rowTiles, 1);

    // ---- layer 2: 256 -> 256, ReLU ----
    zero_f32<<<cdiv(N * 256, T), T, 0, stream>>>(bufT, N * 256);
    spmm_kernel<<<spmmGrid, T, 0, stream>>>(src, dst, ew, bufA, bufT, E, 256);
    gemm_cheb<<<gemmGrid, T, 0, stream>>>(bufA, bufT, (const v16bf*)P2_0, (const v16bf*)P2_1,
                                          b2, bufB, 256, rowTiles, 1);

    // ---- layer 3: 256 -> 256, linear ----
    zero_f32<<<cdiv(N * 256, T), T, 0, stream>>>(bufT, N * 256);
    spmm_kernel<<<spmmGrid, T, 0, stream>>>(src, dst, ew, bufB, bufT, E, 256);
    gemm_cheb<<<gemmGrid, T, 0, stream>>>(bufB, bufT, (const v16bf*)P3_0, (const v16bf*)P3_1,
                                          b3, out, 256, rowTiles, 0);
}
